// MoE_46660524703980
// MI455X (gfx1250) — matrix operand order repaired and checked by reading, not compiled, not run on hardware
//
#include <hip/hip_runtime.h>
#include <hip/hip_bf16.h>
#include <cstdint>
#include <cstddef>

// ---------------- CDNA5 types ----------------
typedef _Float16 half_t;
typedef __attribute__((ext_vector_type(16))) _Float16 v16h;
typedef __attribute__((ext_vector_type(8)))  _Float16 v8h;
typedef __attribute__((ext_vector_type(4)))  _Float16 v4h;
typedef __attribute__((ext_vector_type(8)))  float    v8f;

#define H_DIM 1024
#define F_DIM 4096
#define E_NUM 8

#define MB 32   // token-rows per block
#define NB 64   // output cols per block
#define KB 32   // K chunk (matches wmma f16 K)
#define LDK (KB + 8)  // padded LDS K stride (halves)

// ---------------- zero init ----------------
__global__ void moe_zero_kernel(float* __restrict__ out, long long n, int* __restrict__ cnt) {
  long long i = (long long)blockIdx.x * blockDim.x + threadIdx.x;
  long long stride = (long long)gridDim.x * blockDim.x;
  for (; i < n; i += stride) out[i] = 0.f;
  if (blockIdx.x == 0 && threadIdx.x < E_NUM) cnt[threadIdx.x] = 0;
}

// ---------------- router: logits -> softmax -> top2 -> expert lists ----------------
__global__ __launch_bounds__(64) void moe_router_kernel(
    const float* __restrict__ x, const float* __restrict__ gw,
    float* __restrict__ probs8, int* __restrict__ list,
    float* __restrict__ wslot, int* __restrict__ cnt, int T) {
  const int t = blockIdx.x;
  const int tid = threadIdx.x;
  float acc[E_NUM];
#pragma unroll
  for (int e = 0; e < E_NUM; ++e) acc[e] = 0.f;
  const float* xr = x + (size_t)t * H_DIM;
  for (int i = tid * 4; i < H_DIM; i += 64 * 4) {
    float4 xv = *(const float4*)(xr + i);
#pragma unroll
    for (int e = 0; e < E_NUM; ++e) {
      float4 gv = *(const float4*)(gw + e * H_DIM + i);
      acc[e] += xv.x * gv.x + xv.y * gv.y + xv.z * gv.z + xv.w * gv.w;
    }
  }
  __shared__ float red[64][E_NUM];
#pragma unroll
  for (int e = 0; e < E_NUM; ++e) red[tid][e] = acc[e];
  __syncthreads();
  for (int s = 32; s > 0; s >>= 1) {
    if (tid < s) {
#pragma unroll
      for (int e = 0; e < E_NUM; ++e) red[tid][e] += red[tid + s][e];
    }
    __syncthreads();
  }
  if (tid == 0) {
    float l[E_NUM], p[E_NUM];
    float mx = red[0][0];
#pragma unroll
    for (int e = 0; e < E_NUM; ++e) { l[e] = red[0][e]; mx = fmaxf(mx, l[e]); }
    float s = 0.f;
#pragma unroll
    for (int e = 0; e < E_NUM; ++e) { p[e] = __expf(l[e] - mx); s += p[e]; }
    float inv = 1.f / s;
#pragma unroll
    for (int e = 0; e < E_NUM; ++e) { p[e] *= inv; probs8[(size_t)t * E_NUM + e] = p[e]; }
    // top-2 with lowest-index tie break (JAX top_k)
    int i1 = 0; float b1 = p[0];
#pragma unroll
    for (int e = 1; e < E_NUM; ++e) if (p[e] > b1) { b1 = p[e]; i1 = e; }
    int i2 = -1; float b2 = -1.f;
#pragma unroll
    for (int e = 0; e < E_NUM; ++e) if (e != i1 && p[e] > b2) { b2 = p[e]; i2 = e; }
    // softmax over selected probabilities (b1 >= b2)
    float eb = __expf(b2 - b1);
    float wa = 1.f / (1.f + eb);
    float wb = eb / (1.f + eb);
    int pos1 = atomicAdd(&cnt[i1], 1);
    list[i1 * T + pos1] = t; wslot[i1 * T + pos1] = wa;
    int pos2 = atomicAdd(&cnt[i2], 1);
    list[i2 * T + pos2] = t; wslot[i2 * T + pos2] = wb;
  }
}

// ---------------- aux loss (deterministic fixed-tree reduction) ----------------
__global__ __launch_bounds__(256) void moe_aux_kernel(
    const float* __restrict__ probs8, const int* __restrict__ cnt,
    float* __restrict__ out_aux, int T) {
  __shared__ float red[256][E_NUM];
  float p[E_NUM];
#pragma unroll
  for (int e = 0; e < E_NUM; ++e) p[e] = 0.f;
  for (int t = threadIdx.x; t < T; t += 256) {
#pragma unroll
    for (int e = 0; e < E_NUM; ++e) p[e] += probs8[(size_t)t * E_NUM + e];
  }
#pragma unroll
  for (int e = 0; e < E_NUM; ++e) red[threadIdx.x][e] = p[e];
  __syncthreads();
  for (int s = 128; s > 0; s >>= 1) {
    if (threadIdx.x < s) {
#pragma unroll
      for (int e = 0; e < E_NUM; ++e) red[threadIdx.x][e] += red[threadIdx.x + s][e];
    }
    __syncthreads();
  }
  if (threadIdx.x == 0) {
    float invT = 1.f / (float)T;
    float aux = 0.f;
#pragma unroll
    for (int e = 0; e < E_NUM; ++e)
      aux += ((float)cnt[e] * invT) * (red[0][e] * invT);
    out_aux[0] = aux * (float)E_NUM;
  }
}

// ---------------- pass A: h = silu(x*Wg) * (x*Wu) * w_row  (per expert) ----------------
__global__ __launch_bounds__(256) void moe_gemm_gu_kernel(
    const float* __restrict__ x, const float* __restrict__ Wg,
    const float* __restrict__ Wu, const int* __restrict__ list,
    const float* __restrict__ wslot, const int* __restrict__ cnt,
    half_t* __restrict__ hbuf, int e, int T) {
  const int cn = cnt[e];
  const int m0 = blockIdx.y * MB;
  if (m0 >= cn) return;
  const int n0 = blockIdx.x * NB;
  const int tid  = threadIdx.x;
  const int lane = tid & 31;
  const int w    = tid >> 5;   // wave 0..7
  const int mt   = w >> 2;     // 0..1
  const int nt   = w & 3;      // 0..3
  const int hi   = lane >> 4;
  const int ln   = lane & 15;

  __shared__ half_t Xs[2][MB][LDK];
  __shared__ half_t Gs[2][NB][LDK];
  __shared__ half_t Us[2][NB][LDK];
  __shared__ __align__(16) half_t hs[8][16][16];
  __shared__ int   tokS[MB];
  __shared__ float wS[MB];

  if (tid < MB) {
    int slot = m0 + tid;
    if (slot < cn) { tokS[tid] = list[e * T + slot]; wS[tid] = wslot[e * T + slot]; }
    else           { tokS[tid] = 0;                  wS[tid] = 0.f; }
  }
  __syncthreads();

  // per-thread load coordinates, gather pointer resolved ONCE (kills the
  // per-element tokS->address->load dependency chain seen in the disasm)
  const int xrr = tid >> 3;            // X row 0..31
  const int xcc = (tid & 7) * 4;       // X col quad
  const float* xrow = x + (size_t)tokS[xrr] * H_DIM + xcc;
  const int qn  = (tid & 15) * 4;      // weight col quad
  const int qk0 = tid >> 4;            // weight row 0..15 (and +16)
  const float* gbase = Wg + (size_t)e * H_DIM * F_DIM + n0 + qn;
  const float* ubase = Wu + (size_t)e * H_DIM * F_DIM + n0 + qn;

  auto loadA = [&](int kk, int buf) {
    float4 xv = *(const float4*)(xrow + kk);
    v4h xh = { (half_t)xv.x, (half_t)xv.y, (half_t)xv.z, (half_t)xv.w };
    *(v4h*)&Xs[buf][xrr][xcc] = xh;
    float4 g0 = *(const float4*)(gbase + (size_t)(kk + qk0) * F_DIM);
    float4 g1 = *(const float4*)(gbase + (size_t)(kk + qk0 + 16) * F_DIM);
    float4 u0 = *(const float4*)(ubase + (size_t)(kk + qk0) * F_DIM);
    float4 u1 = *(const float4*)(ubase + (size_t)(kk + qk0 + 16) * F_DIM);
    if (kk + KB < H_DIM) {  // stream next K-chunk (global_prefetch_b8)
      __builtin_prefetch(gbase + (size_t)(kk + KB + qk0) * F_DIM, 0, 1);
      __builtin_prefetch(ubase + (size_t)(kk + KB + qk0) * F_DIM, 0, 1);
    }
    // transposed [n][k] stores for the B-operand layout
    Gs[buf][qn + 0][qk0] = (half_t)g0.x; Gs[buf][qn + 1][qk0] = (half_t)g0.y;
    Gs[buf][qn + 2][qk0] = (half_t)g0.z; Gs[buf][qn + 3][qk0] = (half_t)g0.w;
    Gs[buf][qn + 0][qk0 + 16] = (half_t)g1.x; Gs[buf][qn + 1][qk0 + 16] = (half_t)g1.y;
    Gs[buf][qn + 2][qk0 + 16] = (half_t)g1.z; Gs[buf][qn + 3][qk0 + 16] = (half_t)g1.w;
    Us[buf][qn + 0][qk0] = (half_t)u0.x; Us[buf][qn + 1][qk0] = (half_t)u0.y;
    Us[buf][qn + 2][qk0] = (half_t)u0.z; Us[buf][qn + 3][qk0] = (half_t)u0.w;
    Us[buf][qn + 0][qk0 + 16] = (half_t)u1.x; Us[buf][qn + 1][qk0 + 16] = (half_t)u1.y;
    Us[buf][qn + 2][qk0 + 16] = (half_t)u1.z; Us[buf][qn + 3][qk0 + 16] = (half_t)u1.w;
  };

  v8f cg = {}; v8f cu = {};
  int cur = 0;
  loadA(0, 0);
  for (int kk = 0; kk < H_DIM; kk += KB) {
    __syncthreads();
    if (kk + KB < H_DIM) loadA(kk + KB, cur ^ 1);

    v16h a, bg, bu;
    const half_t* pa0 = &Xs[cur][mt * 16 + ln][16 * hi];
    const half_t* pa1 = &Xs[cur][mt * 16 + ln][16 * hi + 8];
#pragma unroll
    for (int j = 0; j < 8; ++j) { a[j] = pa0[j]; a[8 + j] = pa1[j]; }
    const half_t* pg = &Gs[cur][nt * 16 + ln][16 * hi];
    const half_t* pu = &Us[cur][nt * 16 + ln][16 * hi];
#pragma unroll
    for (int j = 0; j < 16; ++j) { bg[j] = pg[j]; bu[j] = pu[j]; }

    cg = __builtin_amdgcn_wmma_f32_16x16x32_f16(false, a, false, bg, (short)0, cg, false, false);
    cu = __builtin_amdgcn_wmma_f32_16x16x32_f16(false, a, false, bu, (short)0, cu, false, false);
    cur ^= 1;
  }

  // h = silu(g)*u*w_row, staged through LDS for 16B global stores
#pragma unroll
  for (int v = 0; v < 8; ++v) {
    int m = v + 8 * hi;            // row within 16x16 tile
    float g = cg[v], u = cu[v];
    float sig = 1.f / (1.f + __expf(-g));
    hs[w][m][ln] = (half_t)(g * sig * u * wS[mt * 16 + m]);
  }
  __syncthreads();
  {
    int r = lane >> 1, hh = lane & 1;
    v8h val = *(const v8h*)&hs[w][r][8 * hh];
    size_t row = (size_t)(m0 + mt * 16 + r);
    *(v8h*)&hbuf[row * F_DIM + n0 + nt * 16 + 8 * hh] = val;
  }
}

// ---------------- pass B: out[token] += h * Wd  (per expert, atomic scatter) ----------------
__global__ __launch_bounds__(256) void moe_gemm_down_kernel(
    const half_t* __restrict__ hbuf, const float* __restrict__ Wd,
    const int* __restrict__ list, const int* __restrict__ cnt,
    float* __restrict__ out, int e, int T) {
  const int cn = cnt[e];
  const int m0 = blockIdx.y * MB;
  if (m0 >= cn) return;
  const int n0 = blockIdx.x * NB;  // over H
  const int tid  = threadIdx.x;
  const int lane = tid & 31;
  const int w    = tid >> 5;
  const int mt   = w >> 2;
  const int nt   = w & 3;
  const int hi   = lane >> 4;
  const int ln   = lane & 15;

  __shared__ half_t Hs[2][MB][LDK];
  __shared__ half_t Ds[2][NB][LDK];
  __shared__ int tokS[MB];

  if (tid < MB) {
    int slot = m0 + tid;
    tokS[tid] = (slot < cn) ? list[e * T + slot] : 0;
  }
  __syncthreads();

  // h tile: f16 in ws, moved with the CDNA5 async global->LDS path (ASYNCcnt)
  const int hr = tid >> 3;             // 0..31
  const int hc = (tid & 7) * 4;        // quad of halves (8 bytes)
  const half_t* hrow = hbuf + (size_t)(m0 + hr) * F_DIM + hc;
  const int qn  = (tid & 15) * 4;
  const int qk0 = tid >> 4;
  const float* dbase = Wd + (size_t)e * F_DIM * H_DIM + n0 + qn;

  auto loadB = [&](int kk, int buf) {
    uint32_t ldsa = (uint32_t)(uintptr_t)&Hs[buf][hr][hc];
    const half_t* gp = hrow + kk;
    asm volatile("global_load_async_to_lds_b64 %0, %1, off"
                 :: "v"(ldsa), "v"(gp) : "memory");
    float4 d0 = *(const float4*)(dbase + (size_t)(kk + qk0) * H_DIM);
    float4 d1 = *(const float4*)(dbase + (size_t)(kk + qk0 + 16) * H_DIM);
    if (kk + KB < F_DIM)
      __builtin_prefetch(dbase + (size_t)(kk + KB + qk0) * H_DIM, 0, 1);
    Ds[buf][qn + 0][qk0] = (half_t)d0.x; Ds[buf][qn + 1][qk0] = (half_t)d0.y;
    Ds[buf][qn + 2][qk0] = (half_t)d0.z; Ds[buf][qn + 3][qk0] = (half_t)d0.w;
    Ds[buf][qn + 0][qk0 + 16] = (half_t)d1.x; Ds[buf][qn + 1][qk0 + 16] = (half_t)d1.y;
    Ds[buf][qn + 2][qk0 + 16] = (half_t)d1.z; Ds[buf][qn + 3][qk0 + 16] = (half_t)d1.w;
  };

  v8f c = {};
  int cur = 0;
  loadB(0, 0);
  for (int kk = 0; kk < F_DIM; kk += KB) {
    asm volatile("s_wait_asynccnt 0x0" ::: "memory");  // my async h tile landed
    __syncthreads();                                   // ... visible to all waves
    if (kk + KB < F_DIM) loadB(kk + KB, cur ^ 1);

    v16h a, b;
    const half_t* pa0 = &Hs[cur][mt * 16 + ln][16 * hi];
    const half_t* pa1 = &Hs[cur][mt * 16 + ln][16 * hi + 8];
#pragma unroll
    for (int j = 0; j < 8; ++j) { a[j] = pa0[j]; a[8 + j] = pa1[j]; }
    const half_t* pb = &Ds[cur][nt * 16 + ln][16 * hi];
#pragma unroll
    for (int j = 0; j < 16; ++j) b[j] = pb[j];

    c = __builtin_amdgcn_wmma_f32_16x16x32_f16(false, a, false, b, (short)0, c, false, false);
    cur ^= 1;
  }

  // scatter-add: each token gets exactly 2 contributions (commutative -> deterministic)
#pragma unroll
  for (int v = 0; v < 8; ++v) {
    int m = mt * 16 + v + 8 * hi;
    int slot = m0 + m;
    if (slot < cn) {
      int tk = tokS[m];
      atomicAdd(&out[(size_t)tk * H_DIM + n0 + nt * 16 + ln], c[v]);
    }
  }
}

// ---------------- host launcher ----------------
extern "C" void kernel_launch(void* const* d_in, const int* in_sizes, int n_in,
                              void* d_out, int out_size, void* d_ws, size_t ws_size,
                              hipStream_t stream) {
  const float* x  = (const float*)d_in[0];
  const float* gw = (const float*)d_in[1];
  const float* Wg = (const float*)d_in[2];
  const float* Wu = (const float*)d_in[3];
  const float* Wd = (const float*)d_in[4];
  float* out = (float*)d_out;
  const int T = in_sizes[0] / H_DIM;   // B*S tokens
  (void)n_in; (void)out_size; (void)ws_size;

  // workspace layout
  char* ws = (char*)d_ws;
  float*  probs8 = (float*)ws;                                     // T*8 f32
  size_t  o1 = (size_t)T * 8 * 4;
  int*    list   = (int*)(ws + o1);                                // E*T i32
  float*  wslot  = (float*)(ws + o1 + (size_t)E_NUM * T * 4);      // E*T f32
  int*    cnt    = (int*)(ws + o1 + (size_t)E_NUM * T * 8);        // E i32
  size_t  hoff   = (o1 + (size_t)E_NUM * T * 8 + 4096 + 255) & ~(size_t)255;
  half_t* hbuf   = (half_t*)(ws + hoff);                           // T*F f16 (reused per expert)

  long long nout = (long long)T * H_DIM + 1;
  moe_zero_kernel<<<1024, 256, 0, stream>>>(out, nout, cnt);
  moe_router_kernel<<<T, 64, 0, stream>>>(x, gw, probs8, list, wslot, cnt, T);
  moe_aux_kernel<<<1, 256, 0, stream>>>(probs8, cnt, out + (size_t)T * H_DIM, T);

  dim3 blk(256);
  dim3 grA(F_DIM / NB, (T + MB - 1) / MB);  // 64 x 256
  dim3 grB(H_DIM / NB, (T + MB - 1) / MB);  // 16 x 256
  for (int e = 0; e < E_NUM; ++e) {
    moe_gemm_gu_kernel<<<grA, blk, 0, stream>>>(x, Wg, Wu, list, wslot, cnt, hbuf, e, T);
    moe_gemm_down_kernel<<<grB, blk, 0, stream>>>(hbuf, Wd, list, cnt, out, e, T);
  }
}
